// EnhancedProxyNCALoss_3470333575799
// MI455X (gfx1250) — compile-verified, matrix operand order checked by reading, not run on hardware
//
#include <hip/hip_runtime.h>
#include <hip/hip_bf16.h>
#include <math.h>

#define NUM_CLASSES 10000
#define EMBED_DIM   128
#define BATCH       4096
#define SCALE_F     10.0f
#define KSEL        2999             // int((10000-1)*0.3)
#define NTILES      (NUM_CLASSES/16) // 625
#define MTILES      (BATCH/16)       // 256
#define GEMM_WAVES  8
#define ROW_T       256              // threads per row-select block

typedef __attribute__((ext_vector_type(16))) _Float16 v16h;
typedef __attribute__((ext_vector_type(8)))  _Float16 v8h;
typedef __attribute__((ext_vector_type(8)))  float    v8f;

// ---------------------------------------------------------------------------
// Row L2-normalize fp32 -> f16.  One block (128 threads) per row.
// ---------------------------------------------------------------------------
__global__ void normalize_rows_f16(const float* __restrict__ in,
                                   _Float16* __restrict__ out, int rows) {
    int row = blockIdx.x;
    if (row >= rows) return;
    int tid = threadIdx.x; // 0..127 == EMBED_DIM
    float v = in[(size_t)row * EMBED_DIM + tid];
    __shared__ float red[EMBED_DIM];
    red[tid] = v * v;
    __syncthreads();
    #pragma unroll
    for (int off = EMBED_DIM / 2; off > 0; off >>= 1) {
        if (tid < off) red[tid] += red[tid + off];
        __syncthreads();
    }
    float inv = 1.0f / fmaxf(sqrtf(red[0]), 1e-12f);
    out[(size_t)row * EMBED_DIM + tid] = (_Float16)(v * inv);
}

// ---------------------------------------------------------------------------
// sim = (eN @ pN^T) * SCALE  via v_wmma_f32_16x16x32_f16.
// One wave32 per 16x16 tile; 8 waves per block (different N tiles).
// A fragment (16x32 f16): lane l<16 holds row M=l, K={k0..k0+7, k0+16..k0+23};
// lane l>=16 holds row M=l-16, K={k0+8..k0+15, k0+24..k0+31}.  B mirrors A
// (both operands are K-contiguous in memory since B = proxies row-major).
// ---------------------------------------------------------------------------
__global__ void sim_gemm_wmma(const _Float16* __restrict__ eN,
                              const _Float16* __restrict__ pN,
                              float* __restrict__ sim) {
    const int lane  = threadIdx.x;                    // 0..31
    const int wave  = threadIdx.y;                    // 0..7
    const int mtile = blockIdx.x;                     // 0..255
    const int ntile = blockIdx.y * GEMM_WAVES + wave; // class tile
    if (ntile >= NTILES) return;                      // wave-uniform exit

    const int arow = mtile * 16 + (lane & 15);
    const int brow = ntile * 16 + (lane & 15);
    const int ksub = (lane >> 4) * 8;                 // 0 or 8

    const _Float16* aptr = eN + (size_t)arow * EMBED_DIM + ksub;
    const _Float16* bptr = pN + (size_t)brow * EMBED_DIM + ksub;

    v8f c = {};
    #pragma unroll
    for (int k0 = 0; k0 < EMBED_DIM; k0 += 32) {
        if (k0 + 32 < EMBED_DIM) {
            __builtin_prefetch(aptr + k0 + 32, 0, 3); // global_prefetch_b8
            __builtin_prefetch(bptr + k0 + 32, 0, 3);
        }
        v8h alo = *(const v8h*)(aptr + k0);
        v8h ahi = *(const v8h*)(aptr + k0 + 16);
        v8h blo = *(const v8h*)(bptr + k0);
        v8h bhi = *(const v8h*)(bptr + k0 + 16);
        v16h A, B;
        #pragma unroll
        for (int i = 0; i < 8; ++i) {
            A[i] = alo[i]; A[i + 8] = ahi[i];
            B[i] = blo[i]; B[i + 8] = bhi[i];
        }
        // (neg_a, A, neg_b, B, c_mod, C, reuse_a, reuse_b)
        c = __builtin_amdgcn_wmma_f32_16x16x32_f16(false, A, false, B,
                                                   (short)0, c, false, false);
    }

    // C layout: VGPR r, lanes 0-15 -> M=r, N=lane; lanes 16-31 -> M=r+8.
    const int col   = ntile * 16 + (lane & 15);
    const int rbase = mtile * 16 + ((lane >> 4) << 3);
    #pragma unroll
    for (int r = 0; r < 8; ++r)
        sim[(size_t)(rbase + r) * NUM_CLASSES + col] = c[r] * SCALE_F;
}

// ---------------------------------------------------------------------------
// Zero the output accumulator (harness poisons d_out with 0xAA).
// ---------------------------------------------------------------------------
__global__ void zero_out_kernel(float* __restrict__ out, int n) {
    int i = blockIdx.x * blockDim.x + threadIdx.x;
    if (i < n) out[i] = 0.0f;
}

// ---------------------------------------------------------------------------
// Per-row exact top-k softmax focal loss.  One block (256 threads) per row.
// The 40 KB row is staged ONCE into LDS as monotone sort keys; the label slot
// gets sentinel key 0 (global minimum, can never reach top-2999 of 9999).
// All selection passes then run out of LDS:
//   - max over negatives
//   - 4x MSB-first radix histogram -> exact KSEL-th largest value v_k
//   - sum of exp over keys > v_k, with (KSEL - cnt_gt) ties at v_k
// ---------------------------------------------------------------------------
__device__ __forceinline__ unsigned sortKey(float f) {
    unsigned u = __float_as_uint(f);
    return (u & 0x80000000u) ? ~u : (u | 0x80000000u);
}
__device__ __forceinline__ float keyToFloat(unsigned k) {
    unsigned u = (k & 0x80000000u) ? (k & 0x7FFFFFFFu) : ~k;
    return __uint_as_float(u);
}

__global__ void row_select_kernel(const float* __restrict__ sim,
                                  const int* __restrict__ labels,
                                  const float* __restrict__ cw,
                                  float* __restrict__ out) {
    const int row = blockIdx.x;
    const int tid = threadIdx.x; // 0..255
    const float* s = sim + (size_t)row * NUM_CLASSES;
    const int label = labels[row];

    __shared__ unsigned keys[NUM_CLASSES]; // 40 KB, LDS-resident row
    __shared__ unsigned hist[256];
    __shared__ float    red[256];
    __shared__ int      icnt[256];
    __shared__ unsigned sh_prefix;
    __shared__ int      sh_need;

    // ---- fill LDS keys (float4 -> uint4), fuse neg-max ----
    float mloc = -INFINITY;
    const float4* s4 = (const float4*)s; // rows are 16B aligned (40000B stride)
    for (int i = tid; i < NUM_CLASSES / 4; i += ROW_T) {
        float4 v = s4[i];
        int base = i * 4;
        unsigned k0, k1, k2, k3;
        if (base + 0 != label) { k0 = sortKey(v.x); mloc = fmaxf(mloc, v.x); } else k0 = 0u;
        if (base + 1 != label) { k1 = sortKey(v.y); mloc = fmaxf(mloc, v.y); } else k1 = 0u;
        if (base + 2 != label) { k2 = sortKey(v.z); mloc = fmaxf(mloc, v.z); } else k2 = 0u;
        if (base + 3 != label) { k3 = sortKey(v.w); mloc = fmaxf(mloc, v.w); } else k3 = 0u;
        uint4 kk = make_uint4(k0, k1, k2, k3);
        *(uint4*)(&keys[base]) = kk; // ds_store_b128
    }
    red[tid] = mloc;
    __syncthreads();
    #pragma unroll
    for (int off = 128; off > 0; off >>= 1) {
        if (tid < off) red[tid] = fmaxf(red[tid], red[tid + off]);
        __syncthreads();
    }
    const float pos = s[label];
    const float m   = fmaxf(red[0], pos);
    __syncthreads();

    // ---- 4x radix select from LDS: exact KSEL-th largest negative ----
    unsigned prefix = 0;
    int need = KSEL;
    for (int shift = 24; shift >= 0; shift -= 8) {
        unsigned mask = (shift == 24) ? 0u : (0xFFFFFFFFu << (shift + 8));
        hist[tid] = 0;
        __syncthreads();
        for (int c = tid; c < NUM_CLASSES; c += ROW_T) {
            unsigned k = keys[c];
            if ((k & mask) == prefix)
                atomicAdd(&hist[(k >> shift) & 255u], 1u);
        }
        __syncthreads();
        if (tid == 0) {
            int acc = 0, b = 255;
            for (; b > 0; --b) {
                int h = (int)hist[b];
                if (acc + h >= need) break;
                acc += h;
            }
            sh_need   = need - acc;
            sh_prefix = prefix | ((unsigned)b << shift);
        }
        __syncthreads();
        prefix = sh_prefix;
        need   = sh_need;
        __syncthreads();
    }
    const unsigned vk    = prefix;
    const float    vkval = keyToFloat(vk);

    // ---- final pass from LDS: sum exp over strict top (> v_k) ----
    float sumloc = 0.0f;
    int   cntloc = 0;
    for (int c = tid; c < NUM_CLASSES; c += ROW_T) {
        unsigned k = keys[c];
        if (k > vk) { sumloc += expf(keyToFloat(k) - m); ++cntloc; }
    }
    red[tid]  = sumloc;
    icnt[tid] = cntloc;
    __syncthreads();
    #pragma unroll
    for (int off = 128; off > 0; off >>= 1) {
        if (tid < off) { red[tid] += red[tid + off]; icnt[tid] += icnt[tid + off]; }
        __syncthreads();
    }

    if (tid == 0) {
        float S      = red[0] + (float)(KSEL - icnt[0]) * expf(vkval - m);
        float epos   = expf(pos - m);
        float pprob  = epos / (epos + S);
        float ce     = -logf(pprob + 1e-8f);
        float omp    = 1.0f - pprob;
        float focal  = 0.25f * omp * omp * ce;
        float w      = focal * cw[label];
        atomicAdd(out, w * (1.0f / (float)BATCH));
    }
}

// ---------------------------------------------------------------------------
extern "C" void kernel_launch(void* const* d_in, const int* in_sizes, int n_in,
                              void* d_out, int out_size, void* d_ws, size_t ws_size,
                              hipStream_t stream) {
    const float* emb    = (const float*)d_in[0];
    const int*   labels = (const int*)d_in[1];
    const float* cw     = (const float*)d_in[2];
    const float* prox   = (const float*)d_in[3];

    char* ws = (char*)d_ws;
    size_t off = 0;
    _Float16* eN = (_Float16*)(ws + off);
    off += (size_t)BATCH * EMBED_DIM * sizeof(_Float16);        // 1,048,576
    _Float16* pN = (_Float16*)(ws + off);
    off += (size_t)NUM_CLASSES * EMBED_DIM * sizeof(_Float16);  // 2,560,000
    off = (off + 255) & ~(size_t)255;
    float* sim = (float*)(ws + off);                            // 163,840,000

    normalize_rows_f16<<<BATCH, EMBED_DIM, 0, stream>>>(emb, eN, BATCH);
    normalize_rows_f16<<<NUM_CLASSES, EMBED_DIM, 0, stream>>>(prox, pN, NUM_CLASSES);

    dim3 gGrid(MTILES, (NTILES + GEMM_WAVES - 1) / GEMM_WAVES);
    dim3 gBlock(32, GEMM_WAVES);
    sim_gemm_wmma<<<gGrid, gBlock, 0, stream>>>(eN, pN, sim);

    zero_out_kernel<<<(out_size + 255) / 256, 256, 0, stream>>>((float*)d_out, out_size);

    row_select_kernel<<<BATCH, ROW_T, 0, stream>>>(sim, labels, cw, (float*)d_out);
}
